// PPIGraphNet_43276090474889
// MI455X (gfx1250) — matrix operand order repaired and checked by reading, not compiled, not run on hardware
//
#include <hip/hip_runtime.h>
#include <hip/hip_bf16.h>

#define N_NODES 50000
#define N_EDGES 600000
#define G_GRAPHS 64
#define D_HID 128
#define N_HEADS 4
#define C_HEAD 32
#define L_LAYERS 3

typedef __attribute__((ext_vector_type(16))) __bf16 v16bf;
typedef __attribute__((ext_vector_type(8)))  float  v8f;

union FragU {
  v16bf v;
  uint4 q[2];
};

__device__ __forceinline__ unsigned short f2bf(float f) {
  unsigned int u = __float_as_uint(f);
  u += 0x7FFFu + ((u >> 16) & 1u);   // round-to-nearest-even
  return (unsigned short)(u >> 16);
}
__device__ __forceinline__ void atomicMaxF(float* addr, float val) {
  unsigned int* ua = (unsigned int*)addr;
  unsigned int cur = *ua;
  while (__uint_as_float(cur) < val) {
    unsigned int prev = atomicCAS(ua, cur, __float_as_uint(val));
    if (prev == cur) break;
    cur = prev;
  }
}

// ---------------------------------------------------------------------------
// C[M,128] = A[M,128] @ B[128,128] (+bias, optional relu), bf16 in, f32 acc.
// A: row-major bf16, staged per-block into LDS (one coalesced 4KB copy),
//    fragments served as 2x ds_load_b128 per k-step.
// B: pre-packed into WMMA fragment order -> 2x coalesced global_load_b128.
// One 16-row tile per block; 8 waves, wave w owns output cols [16w,16w+16).
// ---------------------------------------------------------------------------
__global__ __launch_bounds__(256) void gemm128_bf16(
    const unsigned short* __restrict__ A,    // [M,128] bf16 bits, row-major
    const unsigned short* __restrict__ Bpk,  // [16384] bf16 bits, fragment-packed
    const float* __restrict__ bias,          // [128] or nullptr
    float* __restrict__ Cf,                  // [M,128] f32 (nullable)
    unsigned short* __restrict__ Cbf,        // [M,128] bf16 (nullable)
    int M, int do_relu)
{
  __shared__ __align__(16) unsigned short As[16 * 128];   // 4KB A tile

  const int tid  = threadIdx.x;
  const int lane = tid & 31;
  const int wave = tid >> 5;
  const int row0 = blockIdx.x << 4;
  const int half = lane >> 4;
  const int l16  = lane & 15;

  // stage A tile: thread t copies 16B: row = t>>4, 16B-chunk = t&15
  {
    int r = tid >> 4, cseg = tid & 15;
    int gr = row0 + r; if (gr >= M) gr = M - 1;
    *(uint4*)&As[r * 128 + cseg * 8] =
        *(const uint4*)(A + (size_t)gr * 128 + cseg * 8);
  }
  __syncthreads();

  v8f acc = {};
#pragma unroll
  for (int t = 0; t < 4; ++t) {          // k0 = 32*t
    const int k0 = t << 5;
    FragU a, b;
    // A fragment: 32 contiguous bytes per lane, k = k0 + 16*half + s (matches packed B)
    a.q[0] = *(const uint4*)&As[l16 * 128 + k0 + 16 * half];
    a.q[1] = *(const uint4*)&As[l16 * 128 + k0 + 16 * half + 8];
    // B fragment: 32 contiguous bytes per lane from packed layout
    const uint4* Bq =
        (const uint4*)(Bpk + (((size_t)(wave * 4 + t) * 32 + lane) << 4));
    b.q[0] = Bq[0];
    b.q[1] = Bq[1];
    acc = __builtin_amdgcn_wmma_f32_16x16x32_bf16(
        false, a.v, false, b.v, (short)0, acc, false, false);
  }

  const int c = (wave << 4) + l16;
  const float bv = bias ? bias[c] : 0.0f;
#pragma unroll
  for (int j = 0; j < 8; ++j) {
    int r = row0 + j + 8 * half;   // D layout: VGPR j -> rows j / j+8
    if (r < M) {
      float v = acc[j] + bv;
      if (do_relu) v = fmaxf(v, 0.0f);
      if (Cf)  Cf [(size_t)r * 128 + c] = v;
      if (Cbf) Cbf[(size_t)r * 128 + c] = f2bf(v);
    }
  }
}

// ---------------------------------------------------------------------------
// Pack a 128x128 f32 weight (row-major K,N) into bf16 WMMA-B fragment order:
// dst[((ct*4 + t)*32 + lane)*16 + s] = src[(t*32 + (lane>>4)*16 + s)*128 + ct*16 + (lane&15)]
__global__ void pack_b_kernel(const float* __restrict__ src,
                              unsigned short* __restrict__ dst) {
  int i = blockIdx.x * blockDim.x + threadIdx.x;
  if (i >= 16384) return;
  int s    = i & 15;
  int lane = (i >> 4) & 31;
  int t    = (i >> 9) & 3;
  int ct   = i >> 11;
  int n = ct * 16 + (lane & 15);
  int k = t * 32 + (lane >> 4) * 16 + s;
  dst[i] = f2bf(src[k * 128 + n]);
}

__global__ void fill_kernel(float* __restrict__ p, float v, size_t n) {
  size_t i = (size_t)blockIdx.x * blockDim.x + threadIdx.x;
  if (i < n) p[i] = v;
}

// h = relu(x @ embed_w + b), K=20 -> plain VALU, also emit bf16 copy
__global__ void embed_kernel(const float* __restrict__ x,
                             const float* __restrict__ w,
                             const float* __restrict__ b,
                             float* __restrict__ h,
                             unsigned short* __restrict__ hbf) {
  int idx = blockIdx.x * blockDim.x + threadIdx.x;
  if (idx >= N_NODES * D_HID) return;
  int n = idx >> 7, d = idx & 127;
  float acc = b[d];
#pragma unroll
  for (int k = 0; k < 20; ++k) acc += x[n * 20 + k] * w[k * 128 + d];
  acc = fmaxf(acc, 0.0f);
  h[idx] = acc;
  hbf[idx] = f2bf(acc);
}

// aew[f,h] = sum_c edge_w[f, h*32+c] * a_edge[h,c]   (per layer, 16 values)
__global__ void aew_kernel(const float* __restrict__ edge_w,
                           const float* __restrict__ a_edge,
                           float* __restrict__ aew) {
  int t = threadIdx.x;
  if (t >= 16) return;
  int f = t >> 2, hh = t & 3;
  float s = 0.f;
  for (int c = 0; c < 32; ++c)
    s += edge_w[f * 128 + hh * 32 + c] * a_edge[hh * 32 + c];
  aew[f * 4 + hh] = s;
}

// alpha_src / alpha_dst per (node, head)
__global__ void alpha_node_kernel(const float* __restrict__ hW,
                                  const float* __restrict__ asrc,
                                  const float* __restrict__ adst,
                                  float* __restrict__ als,
                                  float* __restrict__ ald) {
  int idx = blockIdx.x * blockDim.x + threadIdx.x;
  if (idx >= N_NODES * N_HEADS) return;
  int n = idx >> 2, hh = idx & 3;
  const float* hp = hW + (size_t)n * 128 + hh * 32;
  float s = 0.f, d = 0.f;
#pragma unroll
  for (int c = 0; c < 32; ++c) {
    float v = hp[c];
    s += v * asrc[hh * 32 + c];
    d += v * adst[hh * 32 + c];
  }
  als[idx] = s;
  ald[idx] = d;
}

// pass 1: alpha = leaky_relu(as[src]+ad[dst]+ae), segment max into m[dst]
__global__ void edge_alpha_kernel(const long long* __restrict__ src,
                                  const long long* __restrict__ dst,
                                  const float* __restrict__ edge_attr,
                                  const float* __restrict__ aew,
                                  const float* __restrict__ als,
                                  const float* __restrict__ ald,
                                  float* __restrict__ alpha,
                                  float* __restrict__ m) {
  int idx = blockIdx.x * blockDim.x + threadIdx.x;
  if (idx >= N_EDGES * N_HEADS) return;
  int e = idx >> 2, hh = idx & 3;
  int s = (int)src[e], d = (int)dst[e];
  float4 ea = ((const float4*)edge_attr)[e];
  float ae = ea.x * aew[0 * 4 + hh] + ea.y * aew[1 * 4 + hh] +
             ea.z * aew[2 * 4 + hh] + ea.w * aew[3 * 4 + hh];
  float a = als[s * 4 + hh] + ald[d * 4 + hh] + ae;
  a = (a > 0.f) ? a : 0.2f * a;
  alpha[idx] = a;
  atomicMaxF(&m[d * 4 + hh], a);
}

// pass 2: ex = exp(alpha - m[dst]); denom[dst] += ex (store ex over alpha)
__global__ void edge_exp_kernel(const long long* __restrict__ dst,
                                float* __restrict__ alpha,
                                const float* __restrict__ m,
                                float* __restrict__ denom) {
  int idx = blockIdx.x * blockDim.x + threadIdx.x;
  if (idx >= N_EDGES * N_HEADS) return;
  int e = idx >> 2, hh = idx & 3;
  int d = (int)dst[e];
  float ex = expf(alpha[idx] - m[d * 4 + hh]);
  alpha[idx] = ex;
  atomicAdd(&denom[d * 4 + hh], ex);
}

// pass 3: out[dst] += h[src] * (ex / denom[dst])
__global__ void edge_scatter_kernel(const long long* __restrict__ src,
                                    const long long* __restrict__ dst,
                                    const float* __restrict__ alpha,
                                    const float* __restrict__ denom,
                                    const float* __restrict__ hW,
                                    float* __restrict__ outb) {
  int idx = blockIdx.x * blockDim.x + threadIdx.x;
  if (idx >= N_EDGES * N_HEADS) return;
  int e = idx >> 2, hh = idx & 3;
  int s = (int)src[e], d = (int)dst[e];
  float w = alpha[idx] / (denom[d * 4 + hh] + 1e-16f);
  const float4* hs = (const float4*)(hW + (size_t)s * 128 + hh * 32);
  float* od = outb + (size_t)d * 128 + hh * 32;
#pragma unroll
  for (int j = 0; j < 8; ++j) {
    float4 v = hs[j];
    atomicAdd(&od[4 * j + 0], v.x * w);
    atomicAdd(&od[4 * j + 1], v.y * w);
    atomicAdd(&od[4 * j + 2], v.z * w);
    atomicAdd(&od[4 * j + 3], v.w * w);
  }
}

// h = relu(layernorm(out + bias + h)); one block (128 thr) per node
__global__ __launch_bounds__(128) void ln_kernel(const float* __restrict__ outb,
                                                 const float* __restrict__ bias,
                                                 const float* __restrict__ g,
                                                 const float* __restrict__ bvec,
                                                 float* __restrict__ h,
                                                 unsigned short* __restrict__ hbf) {
  __shared__ float sbuf[128];
  int n = blockIdx.x, d = threadIdx.x;
  size_t idx = (size_t)n * 128 + d;
  float v = outb[idx] + bias[d] + h[idx];
  sbuf[d] = v;
  __syncthreads();
  for (int s = 64; s > 0; s >>= 1) {
    if (d < s) sbuf[d] += sbuf[d + s];
    __syncthreads();
  }
  float mu = sbuf[0] * (1.0f / 128.0f);
  __syncthreads();
  float dv = v - mu;
  sbuf[d] = dv * dv;
  __syncthreads();
  for (int s = 64; s > 0; s >>= 1) {
    if (d < s) sbuf[d] += sbuf[d + s];
    __syncthreads();
  }
  float var = sbuf[0] * (1.0f / 128.0f);
  float y = dv * rsqrtf(var + 1e-5f) * g[d] + bvec[d];
  y = fmaxf(y, 0.0f);
  h[idx] = y;
  hbf[idx] = f2bf(y);
}

// segment mean/max pooling over batch
__global__ void pool_kernel(const float* __restrict__ h,
                            const long long* __restrict__ batch,
                            float* __restrict__ gsum,
                            float* __restrict__ gmax,
                            float* __restrict__ cnt) {
  int idx = blockIdx.x * blockDim.x + threadIdx.x;
  if (idx >= N_NODES * D_HID) return;
  int n = idx >> 7, d = idx & 127;
  int b = (int)batch[n];
  float v = h[idx];
  atomicAdd(&gsum[b * 128 + d], v);
  atomicMaxF(&gmax[b * 128 + d], v);
  if (d == 0) atomicAdd(&cnt[b], 1.0f);
}

// per-graph heads: haddock MLP + interaction + affinity. One block per graph.
__global__ __launch_bounds__(128) void heads_kernel(
    const float* __restrict__ gsum, const float* __restrict__ gmax,
    const float* __restrict__ cnt, const float* __restrict__ hf,
    const float* __restrict__ hw1, const float* __restrict__ hb1,
    const float* __restrict__ hw2, const float* __restrict__ hb2,
    const float* __restrict__ iw1, const float* __restrict__ ib1,
    const float* __restrict__ iw2, const float* __restrict__ ib2,
    const float* __restrict__ iw3, const float* __restrict__ ib3,
    const float* __restrict__ aw1, const float* __restrict__ ab1,
    const float* __restrict__ aw2, const float* __restrict__ ab2,
    const float* __restrict__ aw3, const float* __restrict__ ab3,
    float* __restrict__ out_inter, float* __restrict__ out_aff) {
  __shared__ float comb[384];
  __shared__ float t1[128];
  __shared__ float z1[128];
  __shared__ float z2[64];
  int g = blockIdx.x, d = threadIdx.x;
  float c = cnt[g];
  comb[d]       = gsum[g * 128 + d] / c;
  comb[128 + d] = gmax[g * 128 + d];
  float acc = hb1[d];
#pragma unroll
  for (int f = 0; f < 12; ++f) acc += hf[g * 12 + f] * hw1[f * 128 + d];
  t1[d] = fmaxf(acc, 0.f);
  __syncthreads();
  acc = hb2[d];
  for (int k = 0; k < 128; ++k) acc += t1[k] * hw2[k * 128 + d];
  comb[256 + d] = acc;
  __syncthreads();
  // interaction head
  acc = ib1[d];
  for (int k = 0; k < 384; ++k) acc += comb[k] * iw1[k * 128 + d];
  z1[d] = fmaxf(acc, 0.f);
  __syncthreads();
  if (d < 64) {
    acc = ib2[d];
    for (int k = 0; k < 128; ++k) acc += z1[k] * iw2[k * 64 + d];
    z2[d] = fmaxf(acc, 0.f);
  }
  __syncthreads();
  if (d == 0) {
    acc = ib3[0];
    for (int k = 0; k < 64; ++k) acc += z2[k] * iw3[k];
    out_inter[g] = 1.0f / (1.0f + expf(-acc));
  }
  __syncthreads();
  // affinity head
  acc = ab1[d];
  for (int k = 0; k < 384; ++k) acc += comb[k] * aw1[k * 128 + d];
  z1[d] = fmaxf(acc, 0.f);
  __syncthreads();
  if (d < 64) {
    acc = ab2[d];
    for (int k = 0; k < 128; ++k) acc += z1[k] * aw2[k * 64 + d];
    z2[d] = fmaxf(acc, 0.f);
  }
  __syncthreads();
  if (d == 0) {
    acc = ab3[0];
    for (int k = 0; k < 64; ++k) acc += z2[k] * aw3[k];
    out_aff[g] = acc;
  }
}

// ---------------------------------------------------------------------------
extern "C" void kernel_launch(void* const* d_in, const int* in_sizes, int n_in,
                              void* d_out, int out_size, void* d_ws, size_t ws_size,
                              hipStream_t stream) {
  const float*     x         = (const float*)d_in[0];
  const float*     edge_attr = (const float*)d_in[1];
  const float*     hf        = (const float*)d_in[2];
  const long long* eidx      = (const long long*)d_in[3];
  const long long* batch     = (const long long*)d_in[4];
  const float*     embed_w   = (const float*)d_in[5];
  const float*     embed_b   = (const float*)d_in[6];
  const float*     gat_lin_w = (const float*)d_in[7];
  const float*     gat_asrc  = (const float*)d_in[8];
  const float*     gat_adst  = (const float*)d_in[9];
  const float*     gat_ew    = (const float*)d_in[10];
  const float*     gat_aedge = (const float*)d_in[11];
  const float*     gat_bias  = (const float*)d_in[12];
  const float*     ln_g      = (const float*)d_in[13];
  const float*     ln_b      = (const float*)d_in[14];
  const float*     had_w1    = (const float*)d_in[15];
  const float*     had_b1    = (const float*)d_in[16];
  const float*     had_w2    = (const float*)d_in[17];
  const float*     had_b2    = (const float*)d_in[18];
  const float*     int_w1    = (const float*)d_in[19];
  const float*     int_b1    = (const float*)d_in[20];
  const float*     int_w2    = (const float*)d_in[21];
  const float*     int_b2    = (const float*)d_in[22];
  const float*     int_w3    = (const float*)d_in[23];
  const float*     int_b3    = (const float*)d_in[24];
  const float*     aff_w1    = (const float*)d_in[25];
  const float*     aff_b1    = (const float*)d_in[26];
  const float*     aff_w2    = (const float*)d_in[27];
  const float*     aff_b2    = (const float*)d_in[28];
  const float*     aff_w3    = (const float*)d_in[29];
  const float*     aff_b3    = (const float*)d_in[30];
  const float*     mha_v_w   = (const float*)d_in[31];
  const float*     mha_v_b   = (const float*)d_in[32];
  const float*     mha_o_w   = (const float*)d_in[33];
  const float*     mha_o_b   = (const float*)d_in[34];

  float* outf = (float*)d_out;  // [64 inter][64 aff][N*128 attn][N*128 h]

  const size_t ND = (size_t)N_NODES * D_HID;
  char* base = (char*)d_ws;
  auto carve = [&](size_t bytes) -> void* {
    void* r = (void*)base;
    base += (bytes + 255) & ~(size_t)255;
    return r;
  };
  float*          h      = (float*)carve(ND * 4);
  float*          hW     = (float*)carve(ND * 4);
  float*          outb   = (float*)carve(ND * 4);
  unsigned short* hbf    = (unsigned short*)carve(ND * 2);
  unsigned short* tmpbf  = (unsigned short*)carve(ND * 2);
  float*          alpha  = (float*)carve((size_t)N_EDGES * N_HEADS * 4);
  float*          als    = (float*)carve((size_t)N_NODES * N_HEADS * 4);
  float*          ald    = (float*)carve((size_t)N_NODES * N_HEADS * 4);
  float*          mbuf   = (float*)carve((size_t)N_NODES * N_HEADS * 4);
  float*          den    = (float*)carve((size_t)N_NODES * N_HEADS * 4);
  unsigned short* linwbf = (unsigned short*)carve((size_t)L_LAYERS * 128 * 128 * 2);
  unsigned short* vwbf   = (unsigned short*)carve(128 * 128 * 2);
  unsigned short* owbf   = (unsigned short*)carve(128 * 128 * 2);
  float*          aew    = (float*)carve(16 * 4);
  float*          gsum   = (float*)carve(64 * 128 * 4);
  float*          gmax   = (float*)carve(64 * 128 * 4);
  float*          cnt    = (float*)carve(64 * 4);

  const int TB = 256;
  const unsigned gND = (unsigned)((ND + TB - 1) / TB);
  const unsigned gNH = (unsigned)(((size_t)N_NODES * N_HEADS + TB - 1) / TB);
  const unsigned gEH = (unsigned)(((size_t)N_EDGES * N_HEADS + TB - 1) / TB);
  const long long* esrc = eidx;
  const long long* edst = eidx + N_EDGES;

  // pack weights into bf16 WMMA-B fragment order (once per call)
  for (int l = 0; l < L_LAYERS; ++l)
    pack_b_kernel<<<64, TB, 0, stream>>>(gat_lin_w + (size_t)l * 16384,
                                         linwbf + (size_t)l * 16384);
  pack_b_kernel<<<64, TB, 0, stream>>>(mha_v_w, vwbf);
  pack_b_kernel<<<64, TB, 0, stream>>>(mha_o_w, owbf);

  // node embedding
  embed_kernel<<<gND, TB, 0, stream>>>(x, embed_w, embed_b, h, hbf);

  for (int l = 0; l < L_LAYERS; ++l) {
    fill_kernel<<<gND, TB, 0, stream>>>(outb, 0.f, ND);
    fill_kernel<<<gNH, TB, 0, stream>>>(mbuf, -3.0e38f, (size_t)N_NODES * N_HEADS);
    fill_kernel<<<gNH, TB, 0, stream>>>(den, 0.f, (size_t)N_NODES * N_HEADS);
    aew_kernel<<<1, 32, 0, stream>>>(gat_ew + (size_t)l * 512, gat_aedge + (size_t)l * 128, aew);
    gemm128_bf16<<<N_NODES / 16, 256, 0, stream>>>(
        hbf, linwbf + (size_t)l * 16384, nullptr, hW, nullptr, N_NODES, 0);
    alpha_node_kernel<<<gNH, TB, 0, stream>>>(
        hW, gat_asrc + (size_t)l * 128, gat_adst + (size_t)l * 128, als, ald);
    edge_alpha_kernel<<<gEH, TB, 0, stream>>>(esrc, edst, edge_attr, aew, als, ald, alpha, mbuf);
    edge_exp_kernel<<<gEH, TB, 0, stream>>>(edst, alpha, mbuf, den);
    edge_scatter_kernel<<<gEH, TB, 0, stream>>>(esrc, edst, alpha, den, hW, outb);
    ln_kernel<<<N_NODES, 128, 0, stream>>>(
        outb, gat_bias + (size_t)l * 128, ln_g + (size_t)l * 128, ln_b + (size_t)l * 128, h, hbf);
  }

  // pooling + heads
  fill_kernel<<<32, TB, 0, stream>>>(gsum, 0.f, 64 * 128);
  fill_kernel<<<32, TB, 0, stream>>>(gmax, -3.0e38f, 64 * 128);
  fill_kernel<<<1, 64, 0, stream>>>(cnt, 0.f, 64);
  pool_kernel<<<gND, TB, 0, stream>>>(h, batch, gsum, gmax, cnt);
  heads_kernel<<<G_GRAPHS, 128, 0, stream>>>(
      gsum, gmax, cnt, hf,
      had_w1, had_b1, had_w2, had_b2,
      int_w1, int_b1, int_w2, int_b2, int_w3, int_b3,
      aff_w1, aff_b1, aff_w2, aff_b2, aff_w3, aff_b3,
      outf, outf + 64);

  // MHA (seq_len==1): attn = (h @ v_w + v_b) @ o_w + o_b  -> two WMMA GEMMs
  gemm128_bf16<<<N_NODES / 16, 256, 0, stream>>>(
      hbf, vwbf, mha_v_b, hW, tmpbf, N_NODES, 0);
  gemm128_bf16<<<N_NODES / 16, 256, 0, stream>>>(
      tmpbf, owbf, mha_o_b, outf + 128, nullptr, N_NODES, 0);

  // final h output
  hipMemcpyAsync(outf + 128 + ND, h, ND * sizeof(float),
                 hipMemcpyDeviceToDevice, stream);
}
